// HebbianFastWeights_4750233829415
// MI455X (gfx1250) — matrix operand order repaired and checked by reading, not compiled, not run on hardware
//
#include <hip/hip_runtime.h>

// ---------------------------------------------------------------------------
// Hebbian fast-weights step for MI455X (gfx1250, wave32, WMMA).
//   q/k/v  = x @ W{q,k,v}^T           -> bf16 WMMA GEMMs
//   new_state = 0.9*state + 0.1*k v^T  |  fused, single pass over 134MB state
//   y      = new_state @ q             |  (memory-bound kernel)
//   y_out  = y @ Wout^T + b_out        -> bf16 WMMA GEMM with bias
// ---------------------------------------------------------------------------

typedef __attribute__((ext_vector_type(16))) __bf16 v16bf;
typedef __attribute__((ext_vector_type(8)))  float  v8f;

#define BB   512
#define DD   1024
#define HH   16
#define DHH  64
#define HD   (HH * DHH)      // 1024
#define DECAY_F 0.9f
#define LR_F    0.1f

__device__ __forceinline__ void cvt8(v16bf& d, int base, float4 lo, float4 hi) {
    d[base + 0] = (__bf16)lo.x; d[base + 1] = (__bf16)lo.y;
    d[base + 2] = (__bf16)lo.z; d[base + 3] = (__bf16)lo.w;
    d[base + 4] = (__bf16)hi.x; d[base + 5] = (__bf16)hi.y;
    d[base + 6] = (__bf16)hi.z; d[base + 7] = (__bf16)hi.w;
}

// C(MxN) = A(MxK) @ W(NxK)^T (+ bias[N]).  Row-major A and W, fp32 in memory,
// bf16 in the matrix core.  One wave computes a 16(M) x 64(N) strip: the A
// fragment is loaded once per K-step and reused for 4 WMMA issues.
__global__ __launch_bounds__(128)
void hebb_gemm_nt_wmma(const float* __restrict__ A,
                       const float* __restrict__ W,
                       const float* __restrict__ bias,
                       float* __restrict__ C,
                       int M, int N, int K)
{
    const int lane    = threadIdx.x & 31;
    const int wave    = threadIdx.x >> 5;
    const int tiles_n = N >> 6;                 // 64-wide N tiles
    const int tile    = blockIdx.x * 4 + wave;
    const int m0      = (tile / tiles_n) << 4;
    const int n0      = (tile % tiles_n) << 6;
    if (m0 >= M) return;

    const int row = lane & 15;                  // M row (A) / N col (B,C)
    const int hi  = lane >> 4;                  // half-wave selector

    // A fragment layout (16-bit A 16x32): lanes 0-15 hold K 0..7 / 16..23,
    // lanes 16-31 hold K 8..15 / 24..31.
    const float* ap  = A + (size_t)(m0 + row) * K + hi * 8;
    // B fragment layout (16-bit B 32x16): lane = column; lanes 0-15 hold
    // K 0..7 / 16..23, lanes 16-31 hold K 8..15 / 24..31 (matches A).
    const float* wpb = W + (size_t)(n0 + row) * K + hi * 8;

    v8f acc[4] = {v8f{}, v8f{}, v8f{}, v8f{}};

    for (int k0 = 0; k0 < K; k0 += 32) {
        // Speculative prefetch of the next K-slab (global_prefetch_b8).
        __builtin_prefetch(ap + k0 + 32, 0, 0);

        v16bf a;
        {
            float4 l0 = *(const float4*)(ap + k0);
            float4 l1 = *(const float4*)(ap + k0 + 4);
            float4 h0 = *(const float4*)(ap + k0 + 16);
            float4 h1 = *(const float4*)(ap + k0 + 20);
            cvt8(a, 0, l0, l1);
            cvt8(a, 8, h0, h1);
        }

#pragma unroll
        for (int s = 0; s < 4; ++s) {
            const float* wp = wpb + (size_t)s * 16 * K + k0;
            v16bf b;
            float4 b0 = *(const float4*)(wp);
            float4 b1 = *(const float4*)(wp + 4);
            float4 b2 = *(const float4*)(wp + 16);
            float4 b3 = *(const float4*)(wp + 20);
            cvt8(b, 0, b0, b1);
            cvt8(b, 8, b2, b3);
            acc[s] = __builtin_amdgcn_wmma_f32_16x16x32_bf16(
                false, a, false, b, (short)0, acc[s], false, false);
        }
    }

    // C layout: lane = column, VGPR r holds row (r + 8*hi).
#pragma unroll
    for (int s = 0; s < 4; ++s) {
        const int col  = n0 + s * 16 + row;
        const float bv = bias ? bias[col] : 0.0f;
#pragma unroll
        for (int r = 0; r < 8; ++r) {
            const int m = m0 + r + hi * 8;
            C[(size_t)m * N + col] = acc[s][r] + bv;
        }
    }
}

// One block per (b,h): stream the 64x64 state tile once, write new_state and
// accumulate y = new_state @ q in the same pass (saves a second state read).
__global__ __launch_bounds__(256)
void hebb_state_update(const float* __restrict__ state,
                       const float* __restrict__ q,
                       const float* __restrict__ k,
                       const float* __restrict__ v,
                       float* __restrict__ new_state,
                       float* __restrict__ y)
{
    __shared__ float sq[DHH];
    __shared__ float sk[DHH];
    __shared__ float sv[DHH];

    const int bh  = blockIdx.x;                  // b*H + h
    const int tid = threadIdx.x;

    if (tid < 64)        sq[tid]       = q[bh * DHH + tid];
    else if (tid < 128)  sk[tid - 64]  = k[bh * DHH + tid - 64];
    else if (tid < 192)  sv[tid - 128] = v[bh * DHH + tid - 128];
    __syncthreads();

    const int i  = tid >> 2;            // row 0..63
    const int j0 = (tid & 3) * 16;      // 16-col slice per thread
    const float ki = sk[i] * LR_F;

    const size_t base = (size_t)bh * DHH * DHH + (size_t)i * DHH + j0;
    const float* sp = state + base;
    float*       np = new_state + base;

    float partial = 0.0f;
#pragma unroll
    for (int c = 0; c < 4; ++c) {
        const int j = j0 + c * 4;
        float4 s4 = *(const float4*)(sp + c * 4);
        float4 n4;
        n4.x = DECAY_F * s4.x + ki * sv[j + 0];
        n4.y = DECAY_F * s4.y + ki * sv[j + 1];
        n4.z = DECAY_F * s4.z + ki * sv[j + 2];
        n4.w = DECAY_F * s4.w + ki * sv[j + 3];
        partial += n4.x * sq[j + 0] + n4.y * sq[j + 1]
                 + n4.z * sq[j + 2] + n4.w * sq[j + 3];
        *(float4*)(np + c * 4) = n4;
    }

    // Reduce across the 4 lanes owning row i (lanes 4t..4t+3, same wave32).
    partial += __shfl_xor(partial, 1, 32);
    partial += __shfl_xor(partial, 2, 32);
    if ((tid & 3) == 0) y[bh * DHH + i] = partial;
}

extern "C" void kernel_launch(void* const* d_in, const int* in_sizes, int n_in,
                              void* d_out, int out_size, void* d_ws, size_t ws_size,
                              hipStream_t stream)
{
    (void)in_sizes; (void)n_in; (void)out_size; (void)ws_size;

    const float* x_t   = (const float*)d_in[0];   // (B, D)
    const float* state = (const float*)d_in[1];   // (B, H, DH, DH)
    const float* Wq    = (const float*)d_in[2];   // (H*DH, D)
    const float* Wk    = (const float*)d_in[3];
    const float* Wv    = (const float*)d_in[4];
    const float* Wout  = (const float*)d_in[5];   // (D, H*DH)
    const float* b_out = (const float*)d_in[6];   // (D,)

    float* y_out     = (float*)d_out;                    // (B, D)
    float* new_state = y_out + (size_t)BB * DD;          // (B, H, DH, DH)

    float* q = (float*)d_ws;                             // (B, H*DH)
    float* k = q + (size_t)BB * HD;
    float* v = k + (size_t)BB * HD;
    float* y = v + (size_t)BB * HD;                      // (B, H*DH)

    // (M/16)*(N/64) wave-tiles, 4 waves per 128-thread block.
    const int gemm_blocks = (BB / 16) * (DD / 64) / 4;   // 128

    hebb_gemm_nt_wmma<<<gemm_blocks, 128, 0, stream>>>(x_t, Wq, nullptr, q, BB, HD, DD);
    hebb_gemm_nt_wmma<<<gemm_blocks, 128, 0, stream>>>(x_t, Wk, nullptr, k, BB, HD, DD);
    hebb_gemm_nt_wmma<<<gemm_blocks, 128, 0, stream>>>(x_t, Wv, nullptr, v, BB, HD, DD);

    hebb_state_update<<<BB * HH, 256, 0, stream>>>(state, q, k, v, new_state, y);

    hebb_gemm_nt_wmma<<<gemm_blocks, 128, 0, stream>>>(y, Wout, b_out, y_out, BB, DD, HD);
}
